// AttentionHead_29016799052430
// MI455X (gfx1250) — matrix operand order repaired and checked by reading, not compiled, not run on hardware
//
#include <hip/hip_runtime.h>
#include <hip/hip_bf16.h>
#include <stdint.h>

#define B_  16
#define T_  2048
#define C_  288
#define HS  32

typedef __attribute__((ext_vector_type(16))) __bf16        v16bf;
typedef __attribute__((ext_vector_type(8)))  float         v8f;
typedef __attribute__((ext_vector_type(8)))  unsigned int  v8u;

__device__ __forceinline__ unsigned short f2bf(float f) {
  unsigned int u = __float_as_uint(f);
  u += 0x7FFFu + ((u >> 16) & 1u);          // round-to-nearest-even
  return (unsigned short)(u >> 16);
}

// Cross half-wave exchange (lane ^ 16) via v_permlanex16_b32: identity lane
// selects make each lane read the same lane index of the other 16-lane row.
// Pure VALU -- no LDS round trip, no DScnt wait.
__device__ __forceinline__ int xor16_i(int v) {
  return __builtin_amdgcn_permlanex16(v, v, 0x76543210, (int)0xfedcba98,
                                      false, false);
}
__device__ __forceinline__ float xor16_f(float v) {
  return __uint_as_float((unsigned int)xor16_i((int)__float_as_uint(v)));
}

// Pack two positive f32 -> bf16x2 {lo=a, hi=b} by truncation, one v_perm_b32:
// bytes {b[3], b[2], a[3], a[2]}.
__device__ __forceinline__ unsigned int pack_bf16_trunc(float a, float b) {
  return __builtin_amdgcn_perm(__float_as_uint(b), __float_as_uint(a),
                               0x07060302u);
}

// ---------------------------------------------------------------------------
// Projection: k = (x@Wk) * hs^-0.5 * log2(e)   [exp2-domain scores]
//             q = x@Wq,  vT = (x@Wv)^T          (fp32 acc -> bf16 RNE)
// One wave per (b,t) row; lane j owns output column j. x[c] broadcasts within
// the wave (one transaction), W loads are fully coalesced across lanes.
// ---------------------------------------------------------------------------
__global__ __launch_bounds__(256) void proj_kernel(
    const float* __restrict__ x,  const float* __restrict__ Wk,
    const float* __restrict__ Wq, const float* __restrict__ Wv,
    unsigned short* __restrict__ kb, unsigned short* __restrict__ qb,
    unsigned short* __restrict__ vT) {
  const int row = blockIdx.x * 8 + (threadIdx.x >> 5);   // row = b*T + t
  const int j   = threadIdx.x & 31;
  const float* xr = x + (size_t)row * C_;
  float ak = 0.f, aq = 0.f, av = 0.f;
#pragma unroll 4
  for (int c = 0; c < C_; ++c) {
    const float xv = xr[c];
    ak = fmaf(xv, Wk[c * HS + j], ak);
    aq = fmaf(xv, Wq[c * HS + j], aq);
    av = fmaf(xv, Wv[c * HS + j], av);
  }
  // fold hs^-0.5 AND log2(e) into K so softmax runs in pure exp2 domain
  ak *= 0.17677669529663687f * 1.4426950408889634f;
  kb[(size_t)row * HS + j] = f2bf(ak);
  qb[(size_t)row * HS + j] = f2bf(aq);
  const int b = row / T_, t = row - b * T_;
  vT[((size_t)b * HS + j) * T_ + t] = f2bf(av);  // transposed V for A-fragments
}

// ---------------------------------------------------------------------------
// Flash attention, one wave per 16-row t-tile.
//   S^T = Q * K^T            (v_wmma_f32_16x16x32_bf16, x2 per 32-wide s-step)
//   O^T += V^T * P^T         (v_wmma_f32_16x16x32_bf16, x2)
// Online softmax in exp2 domain: per-t row state lives in lane pair (n, n+16);
// all row reductions are a single permlanex16.
// ---------------------------------------------------------------------------
__global__ __launch_bounds__(128) void attn_kernel(
    const unsigned short* __restrict__ kb,
    const unsigned short* __restrict__ qb,
    const unsigned short* __restrict__ vT,
    float* __restrict__ out) {
  const int lane  = threadIdx.x & 31;
  const int wave  = threadIdx.x >> 5;
  const int tile  = blockIdx.x * 4 + wave;       // global t-tile id
  const int b     = tile >> 7;                   // / (T_/16)
  const int t0    = (tile & 127) << 4;
  const int tn    = lane & 15;
  const bool hi   = lane >= 16;
  const int klo   = hi ? 8 : 0;                  // A-frag / D-tile K half offset
  const int tglob = t0 + tn;

  const unsigned short* kbB = kb + (size_t)b * T_ * HS;
  const unsigned short* qbB = qb + (size_t)b * T_ * HS;
  const unsigned short* vTB = vT + (size_t)b * HS * T_;

  // Loop-invariant B-fragment: K^T of this t-tile. B[h, n] = k[t0+n, h].
  // VGPR0-3: K = klo..klo+7, VGPR4-7: K = 16+klo..23+klo.
  v8u bku;
  {
    const unsigned short* kr = kbB + (size_t)(t0 + tn) * HS + klo;
    uint4 p = *(const uint4*)kr, q4 = *(const uint4*)(kr + 16);
    bku[0]=p.x; bku[1]=p.y; bku[2]=p.z; bku[3]=p.w;
    bku[4]=q4.x; bku[5]=q4.y; bku[6]=q4.z; bku[7]=q4.w;
  }
  const v16bf Bk = __builtin_bit_cast(v16bf, bku);

  v8f O0 = {}; v8f O1 = {};
  float m = -INFINITY, l = 0.f;

  const int nsteps = (t0 >> 5) + 1;              // only the last step is masked

  for (int js = 0; js < nsteps; ++js) {
    const int s0 = js << 5;

    // Q A-fragments (rows s0..s0+15 and s0+16..s0+31).
    // Lane: M = tn, VGPR0-3: K = klo..klo+7, VGPR4-7: K = 16+klo..23+klo.
    v8u a0u, a1u, av0u, av1u;
    {
      const unsigned short* qr = qbB + (size_t)(s0 + tn) * HS + klo;
      uint4 p = *(const uint4*)qr, q4 = *(const uint4*)(qr + 16);
      a0u[0]=p.x; a0u[1]=p.y; a0u[2]=p.z; a0u[3]=p.w;
      a0u[4]=q4.x; a0u[5]=q4.y; a0u[6]=q4.z; a0u[7]=q4.w;
      const unsigned short* qr1 = qr + 16 * HS;
      uint4 r = *(const uint4*)qr1, s = *(const uint4*)(qr1 + 16);
      a1u[0]=r.x; a1u[1]=r.y; a1u[2]=r.z; a1u[3]=r.w;
      a1u[4]=s.x; a1u[5]=s.y; a1u[6]=s.z; a1u[7]=s.w;
    }
    // V^T A-fragments: tile0 rows h=0..15, tile1 rows h=16..31, K = s_rel.
    {
      const unsigned short* vr = vTB + (size_t)tn * T_ + s0 + (hi ? 16 : 0);
      uint4 p = *(const uint4*)vr, q4 = *(const uint4*)(vr + 8);
      av0u[0]=p.x; av0u[1]=p.y; av0u[2]=p.z; av0u[3]=p.w;
      av0u[4]=q4.x; av0u[5]=q4.y; av0u[6]=q4.z; av0u[7]=q4.w;
      const unsigned short* vr1 = vr + 16 * T_;
      uint4 r = *(const uint4*)vr1, s = *(const uint4*)(vr1 + 8);
      av1u[0]=r.x; av1u[1]=r.y; av1u[2]=r.z; av1u[3]=r.w;
      av1u[4]=s.x; av1u[5]=s.y; av1u[6]=s.z; av1u[7]=s.w;
    }

    // Score tiles S^T (M = s_rel, N = t_rel); k pre-scaled by hs^-0.5*log2e.
    const v8f zero = {};
    v8f D0 = __builtin_amdgcn_wmma_f32_16x16x32_bf16(
        false, __builtin_bit_cast(v16bf, a0u), false, Bk, (short)0, zero, false, false);
    v8f D1 = __builtin_amdgcn_wmma_f32_16x16x32_bf16(
        false, __builtin_bit_cast(v16bf, a1u), false, Bk, (short)0, zero, false, false);

    if (js == nsteps - 1) {                      // causal mask: s > t -> -inf
#pragma unroll
      for (int r = 0; r < 8; ++r) {
        const int sg = s0 + klo + r;             // D VGPR r holds M = klo + r
        if (sg > tglob)      D0[r] = -1e30f;
        if (sg + 16 > tglob) D1[r] = -1e30f;
      }
    }

    // ---- online softmax, exp2 domain (row state shared via permlanex16) ----
    float tmax = D0[0];
#pragma unroll
    for (int r = 1; r < 8; ++r) tmax = fmaxf(tmax, D0[r]);
#pragma unroll
    for (int r = 0; r < 8; ++r) tmax = fmaxf(tmax, D1[r]);
    tmax = fmaxf(tmax, xor16_f(tmax));
    const float mnew = fmaxf(m, tmax);
    const float corr = __builtin_amdgcn_exp2f(m - mnew);

    float p0[8], p1[8], psum = 0.f;
#pragma unroll
    for (int r = 0; r < 8; ++r) {
      p0[r] = __builtin_amdgcn_exp2f(D0[r] - mnew);
      p1[r] = __builtin_amdgcn_exp2f(D1[r] - mnew);
      psum += p0[r] + p1[r];
    }
    psum += xor16_f(psum);
    l = l * corr + psum;
    m = mnew;
#pragma unroll
    for (int r = 0; r < 8; ++r) { O0[r] *= corr; O1[r] *= corr; }

    // ---- P^T as B-fragment: one lane-pair exchange per packed register ----
    unsigned int pk0[4], pk1[4];
#pragma unroll
    for (int i = 0; i < 4; ++i) {
      pk0[i] = pack_bf16_trunc(p0[2*i], p0[2*i+1]);
      pk1[i] = pack_bf16_trunc(p1[2*i], p1[2*i+1]);
    }
    v8u bpu;
#pragma unroll
    for (int i = 0; i < 4; ++i) {
      const unsigned int x0 = (unsigned int)xor16_i((int)pk1[i]);
      const unsigned int x1 = (unsigned int)xor16_i((int)pk0[i]);
      bpu[i]     = hi ? x0     : pk0[i];   // K = 2i   (low) / 16+2i (high)
      bpu[4 + i] = hi ? pk1[i] : x1;       // K = 8+2i (low) / 24+2i (high)
    }
    const v16bf Bp = __builtin_bit_cast(v16bf, bpu);

    // O^T += V^T * P^T  (two h-halves)
    O0 = __builtin_amdgcn_wmma_f32_16x16x32_bf16(
        false, __builtin_bit_cast(v16bf, av0u), false, Bp, (short)0, O0, false, false);
    O1 = __builtin_amdgcn_wmma_f32_16x16x32_bf16(
        false, __builtin_bit_cast(v16bf, av1u), false, Bp, (short)0, O1, false, false);
  }

  // Epilogue: out[b, t, h] = O^T[h, t] / l
  const float rl = 1.0f / l;
#pragma unroll
  for (int r = 0; r < 8; ++r) { O0[r] *= rl; O1[r] *= rl; }
  float* orow = out + ((size_t)(b * T_ + tglob)) * HS + klo;
  *(float4*)(orow)      = make_float4(O0[0], O0[1], O0[2], O0[3]);
  *(float4*)(orow + 4)  = make_float4(O0[4], O0[5], O0[6], O0[7]);
  *(float4*)(orow + 16) = make_float4(O1[0], O1[1], O1[2], O1[3]);
  *(float4*)(orow + 20) = make_float4(O1[4], O1[5], O1[6], O1[7]);
}

// ---------------------------------------------------------------------------
extern "C" void kernel_launch(void* const* d_in, const int* in_sizes, int n_in,
                              void* d_out, int out_size, void* d_ws, size_t ws_size,
                              hipStream_t stream) {
  const float* x  = (const float*)d_in[0];
  const float* Wk = (const float*)d_in[1];
  const float* Wq = (const float*)d_in[2];
  const float* Wv = (const float*)d_in[3];

  unsigned short* kb = (unsigned short*)d_ws;                 // [B,T,32] bf16
  unsigned short* qb = kb + (size_t)B_ * T_ * HS;             // [B,T,32] bf16
  unsigned short* vT = qb + (size_t)B_ * T_ * HS;             // [B,32,T] bf16

  proj_kernel<<<(B_ * T_) / 8, 256, 0, stream>>>(x, Wk, Wq, Wv, kb, qb, vT);
  attn_kernel<<<(B_ * (T_ / 16)) / 4, 128, 0, stream>>>(kb, qb, vT, (float*)d_out);
}
